// MultiHeadAttention_4750233830137
// MI455X (gfx1250) — compile-verified, matrix operand order checked by reading, not run on hardware
//
#include <hip/hip_runtime.h>
#include <hip/hip_bf16.h>

typedef __attribute__((ext_vector_type(16))) _Float16 v16h;
typedef __attribute__((ext_vector_type(8)))  _Float16 v8h;
typedef __attribute__((ext_vector_type(8)))  float    v8f;

constexpr int Bn  = 4;
constexpr int Ln  = 2048;
constexpr int Dn  = 1024;
constexpr int Hn  = 16;
constexpr int DHn = 64;
constexpr int Mn  = Bn * Ln;          // 8192 rows

// ---------------------------------------------------------------------------
// Fragment helpers (CDNA5 WMMA 16x16x32 f16 layout, wave32):
// A (16xK): lane holds row m=lane&15; halves 0..7 -> K=kb..kb+7,
//           halves 8..15 -> K=kb+16..kb+23, kb = (lane>=16)?8:0.
// B loaded identically from Bt[N][K] (B stored K-contiguous per column).
// C/D (f32): lane holds col n=lane&15; vgpr r -> row r + ((lane>=16)?8:0).
// ---------------------------------------------------------------------------
__device__ inline v16h load_frag(const _Float16* p) {
  v8h lo = *(const v8h*)(p);
  v8h hi = *(const v8h*)(p + 16);
  return __builtin_shufflevector(lo, hi, 0,1,2,3,4,5,6,7,8,9,10,11,12,13,14,15);
}

__device__ inline v8f wmma16(v16h a, v16h b, v8f c) {
  return __builtin_amdgcn_wmma_f32_16x16x32_f16(false, a, false, b, (short)0, c,
                                                false, false);
}

// CDNA5 async copy: global memory -> LDS, 16 bytes per lane, no VGPR data.
// Tracked by ASYNCcnt (CDNA5 ISA).
__device__ inline void async_copy_b128(unsigned lds_off, const void* gptr) {
  asm volatile("global_load_async_to_lds_b128 %0, %1, off"
               :: "v"(lds_off), "v"(gptr)
               : "memory");
}
__device__ inline void wait_async_all() {
  asm volatile("s_wait_asynccnt 0" ::: "memory");
}

// ---------------------------------------------------------------------------
// Weight convert: Wt[n*D + k] = (f16) W[k*D + n]   (transpose so K contiguous)
// ---------------------------------------------------------------------------
__global__ __launch_bounds__(256) void convert_wt_kernel(
    const float* __restrict__ W, _Float16* __restrict__ Wt) {
  int idx = blockIdx.x * 256 + threadIdx.x;   // idx = n*D + k
  int n = idx >> 10;
  int kk = idx & (Dn - 1);
  Wt[idx] = (_Float16)W[kk * Dn + n];
}

// ---------------------------------------------------------------------------
// LayerNorm. If attn != nullptr: val = x + attn, residual stored to resid_out.
// ---------------------------------------------------------------------------
__global__ __launch_bounds__(256) void ln_kernel(
    const float* __restrict__ x, const float* __restrict__ attn,
    const float* __restrict__ g, const float* __restrict__ b,
    float* __restrict__ resid_out, _Float16* __restrict__ out) {
  __shared__ float red[256];
  const int row = blockIdx.x;
  const int t = threadIdx.x;

  float4 v = ((const float4*)(x + (size_t)row * Dn))[t];
  if (attn != nullptr) {
    float4 av = ((const float4*)(attn + (size_t)row * Dn))[t];
    v.x += av.x; v.y += av.y; v.z += av.z; v.w += av.w;
    ((float4*)(resid_out + (size_t)row * Dn))[t] = v;
  }

  red[t] = v.x + v.y + v.z + v.w;
  __syncthreads();
  for (int o = 128; o > 0; o >>= 1) {
    if (t < o) red[t] += red[t + o];
    __syncthreads();
  }
  const float mu = red[0] * (1.0f / Dn);
  __syncthreads();

  const float dx = v.x - mu, dy = v.y - mu, dz = v.z - mu, dw = v.w - mu;
  red[t] = dx * dx + dy * dy + dz * dz + dw * dw;
  __syncthreads();
  for (int o = 128; o > 0; o >>= 1) {
    if (t < o) red[t] += red[t + o];
    __syncthreads();
  }
  const float rstd = rsqrtf(red[0] * (1.0f / Dn) + 1e-6f);

  const float4 gv = ((const float4*)g)[t];
  const float4 bv = ((const float4*)b)[t];
  _Float16* o4 = out + (size_t)row * Dn + t * 4;
  o4[0] = (_Float16)(dx * rstd * gv.x + bv.x);
  o4[1] = (_Float16)(dy * rstd * gv.y + bv.y);
  o4[2] = (_Float16)(dz * rstd * gv.z + bv.z);
  o4[3] = (_Float16)(dw * rstd * gv.w + bv.w);
}

// ---------------------------------------------------------------------------
// WMMA GEMM: C[8192,1024] = A_f16[M,1024] x Bt_f16[N,1024]^T + bias.
// Block = 8 waves; wave computes 32 rows x 64 cols (8 WMMA / k-step, each
// LDS B fragment reused twice). The shared 64x32 B tile is staged via async
// global->LDS copies, double buffered (ASYNCcnt). K-loop manually unrolled
// 2x with statically-named parity registers (no accumulator phi-copies).
// LDS rows padded to 40 halves (20 DWORDs) -> conflict-free ds_load_b128.
// mode 0: f16 [B,H,L,DH] (Q/K)   mode 2: f16 [B,H,DH,L] (V^T)
// mode 3: exact GELU -> f16 row-major   mode 4: +resid -> f32 row-major
// ---------------------------------------------------------------------------
__global__ __launch_bounds__(256) void gemm_wmma_kernel(
    const _Float16* __restrict__ A, const _Float16* __restrict__ Bt,
    const float* __restrict__ bias, const float* __restrict__ resid,
    void* __restrict__ out, int mode) {
  constexpr int BP = 40;                       // padded row stride (halves)
  __shared__ _Float16 btile[2][64 * BP];       // 2 x 5 KB double buffer

  const int tid  = threadIdx.x;
  const int lane = tid & 31;
  const int wave = tid >> 5;

  const int n4     = blockIdx.x >> 5;          // 0..15  -> 64-col strip
  const int mgroup = blockIdx.x & 31;          // 0..31  -> 8 waves * 32 rows
  const int m0 = (mgroup * 8 + wave) * 32;
  const int n0 = n4 * 64;

  const int mr = lane & 15;
  const int kb = (lane >> 4) << 3;             // 0 or 8

  // Async-copy assignment: thread t copies 16B of row r (= rel. B column).
  const int r  = tid >> 2;                     // 0..63
  const int cc = (tid & 3) * 8;                // 0,8,16,24 halves
  const _Float16* gsrc = Bt + (size_t)(n0 + r) * Dn + cc;
  const unsigned lds0 = (unsigned)(uintptr_t)&btile[0][r * BP + cc];
  const unsigned lds1 = (unsigned)(uintptr_t)&btile[1][r * BP + cc];

  const _Float16* ar0 = A + (size_t)(m0 + mr) * Dn + kb;
  const _Float16* ar1 = A + (size_t)(m0 + 16 + mr) * Dn + kb;

  // Prologue: stage k-tile 0 into buf0, load first A fragments.
  async_copy_b128(lds0, gsrc);
  v16h a00 = load_frag(ar0);
  v16h a01 = load_frag(ar1);
  v8f acc0[4] = {};
  v8f acc1[4] = {};
  wait_async_all();
  __syncthreads();

  for (int kt = 0; kt < 32; kt += 2) {
    // ---- even half: compute buf0, stage kt+1 -> buf1 -------------------
    async_copy_b128(lds1, gsrc + (kt + 1) * 32);
    v16h a10 = load_frag(ar0 + (kt + 1) * 32);
    v16h a11 = load_frag(ar1 + (kt + 1) * 32);
    {
      const _Float16* bl = &btile[0][0];
#pragma unroll
      for (int j = 0; j < 4; ++j) {
        v16h bb = load_frag(bl + (j * 16 + mr) * BP + kb);  // ds_load_b128
        acc0[j] = wmma16(a00, bb, acc0[j]);
        acc1[j] = wmma16(a01, bb, acc1[j]);
      }
    }
    wait_async_all();
    __syncthreads();

    // ---- odd half: compute buf1, stage kt+2 -> buf0 --------------------
    if (kt + 2 < 32) {
      async_copy_b128(lds0, gsrc + (kt + 2) * 32);
      a00 = load_frag(ar0 + (kt + 2) * 32);
      a01 = load_frag(ar1 + (kt + 2) * 32);
    }
    {
      const _Float16* bl = &btile[1][0];
#pragma unroll
      for (int j = 0; j < 4; ++j) {
        v16h bb = load_frag(bl + (j * 16 + mr) * BP + kb);
        acc0[j] = wmma16(a10, bb, acc0[j]);
        acc1[j] = wmma16(a11, bb, acc1[j]);
      }
    }
    if (kt + 2 < 32) wait_async_all();
    __syncthreads();
  }

  // Epilogue.
  const int rowoff = (lane >> 4) << 3;
#pragma unroll
  for (int i = 0; i < 2; ++i) {
    const v8f* accp = i ? acc1 : acc0;
#pragma unroll
    for (int j = 0; j < 4; ++j) {
      const int n = n0 + j * 16 + (lane & 15);
      const float bsv = bias[n];
#pragma unroll
      for (int rr = 0; rr < 8; ++rr) {
        const int m = m0 + i * 16 + rowoff + rr;
        float c = accp[j][rr] + bsv;
        if (mode == 0) {
          const int b_ = m >> 11, l = m & (Ln - 1), h = n >> 6, dh = n & 63;
          ((_Float16*)out)[(((size_t)(b_ * Hn + h) * Ln + l) * DHn) + dh] = (_Float16)c;
        } else if (mode == 2) {
          const int b_ = m >> 11, l = m & (Ln - 1), h = n >> 6, dh = n & 63;
          ((_Float16*)out)[(((size_t)(b_ * Hn + h) * DHn + dh) * Ln) + l] = (_Float16)c;
        } else if (mode == 3) {
          const float ge = 0.5f * c * (1.0f + erff(c * 0.70710678118654752f));
          ((_Float16*)out)[(size_t)m * Dn + n] = (_Float16)ge;
        } else {  // mode 4
          ((float*)out)[(size_t)m * Dn + n] = c + resid[(size_t)m * Dn + n];
        }
      }
    }
  }
}

// ---------------------------------------------------------------------------
// Attention: one workgroup (8 waves) per (b, h, 16-row q tile).
// LDS: f32 scores[16][2048] | f16 probs[16][2048] | f32 out[16][64] | stats.
// ---------------------------------------------------------------------------
constexpr int SCORES_OFF = 0;                            // 131072 B
constexpr int PROBS_OFF  = 16 * Ln * 4;                  // + 65536 B
constexpr int OUT_OFF    = PROBS_OFF + 16 * Ln * 2;      // + 4096 B
constexpr int PART_OFF   = OUT_OFF + 16 * 64 * 4;        // + 1024 B
constexpr int RMAX_OFF   = PART_OFF + 16 * 16 * 4;       // + 64 B
constexpr int RSUM_OFF   = RMAX_OFF + 16 * 4;            // + 64 B
constexpr int SMEM_BYTES = RSUM_OFF + 16 * 4;            // ~201 KB (<320KB WGP LDS)

__global__ __launch_bounds__(256) void attn_kernel(
    const _Float16* __restrict__ q, const _Float16* __restrict__ k,
    const _Float16* __restrict__ vT, const int* __restrict__ mask,
    float* __restrict__ attn_out) {
  extern __shared__ char smem[];
  float*    scores = (float*)(smem + SCORES_OFF);
  _Float16* probs  = (_Float16*)(smem + PROBS_OFF);
  float*    outbuf = (float*)(smem + OUT_OFF);
  float*    part   = (float*)(smem + PART_OFF);
  float*    rmax   = (float*)(smem + RMAX_OFF);
  float*    rsum   = (float*)(smem + RSUM_OFF);

  const int tid  = threadIdx.x;
  const int lane = tid & 31;
  const int wave = tid >> 5;

  const int bh    = blockIdx.x >> 7;   // b*16 + h   (L/16 = 128 q-tiles)
  const int qtile = blockIdx.x & 127;
  const int b_    = bh >> 4;
  const int q0    = qtile * 16;

  const int mr = lane & 15;
  const int kb = (lane >> 4) << 3;
  const int rowoff = (lane >> 4) << 3;

  // Q fragments: registers, reused for all 16 key tiles of this wave.
  const _Float16* qrow = q + ((size_t)bh * Ln + q0 + mr) * DHn + kb;
  const v16h aq0 = load_frag(qrow);
  const v16h aq1 = load_frag(qrow + 32);

  // Emit one 16x16 score tile (already accumulated in c) into LDS.
  auto emit_scores = [&](int key0, const v8f& c) {
    const int n = key0 + (lane & 15);
    const float maskadd = (1.0f - (float)mask[b_ * Ln + n]) * -1e30f;
#pragma unroll
    for (int rr = 0; rr < 8; ++rr)
      scores[(rowoff + rr) * Ln + n] = c[rr] * 0.125f + maskadd;
  };

  // ---- Phase 1: scores = Q K^T / sqrt(DH) + mask; 2x-unrolled pipeline --
  const int keybase = wave * 256;                // 256 keys per wave
  {
    const _Float16* kbase = k + ((size_t)bh * Ln + keybase + mr) * DHn + kb;
    constexpr int KTS = 16 * DHn;                // halves per 16-key tile
    v16h c0a = load_frag(kbase);
    v16h c0b = load_frag(kbase + 32);
    for (int kt = 0; kt < 16; kt += 2) {
      v16h c1a = load_frag(kbase + (kt + 1) * KTS);
      v16h c1b = load_frag(kbase + (kt + 1) * KTS + 32);
      {
        v8f c = {};
        c = wmma16(aq0, c0a, c);
        c = wmma16(aq1, c0b, c);
        emit_scores(keybase + kt * 16, c);
      }
      if (kt + 2 < 16) {
        c0a = load_frag(kbase + (kt + 2) * KTS);
        c0b = load_frag(kbase + (kt + 2) * KTS + 32);
      }
      {
        v8f c = {};
        c = wmma16(aq0, c1a, c);
        c = wmma16(aq1, c1b, c);
        emit_scores(keybase + (kt + 1) * 16, c);
      }
    }
  }
  __syncthreads();

  // ---- Phase 2: softmax (16 rows x 2048, 16 threads per row) -----------
  const int srow   = tid >> 4;   // 0..15
  const int schunk = tid & 15;   // 128 cols each
  const float* sr = scores + srow * Ln + schunk * 128;
  float mx = -3.0e38f;
  for (int i = 0; i < 128; ++i) mx = fmaxf(mx, sr[i]);
  part[srow * 16 + schunk] = mx;
  __syncthreads();
  if (tid < 16) {
    float m = part[tid * 16];
    for (int i = 1; i < 16; ++i) m = fmaxf(m, part[tid * 16 + i]);
    rmax[tid] = m;
  }
  __syncthreads();

  const float rm = rmax[srow];
  _Float16* pr = probs + srow * Ln + schunk * 128;
  float sum = 0.0f;
  for (int i = 0; i < 128; ++i) {
    const float e = __expf(sr[i] - rm);
    pr[i] = (_Float16)e;
    sum += e;
  }
  part[srow * 16 + schunk] = sum;
  __syncthreads();
  if (tid < 16) {
    float s = 0.0f;
    for (int i = 0; i < 16; ++i) s += part[tid * 16 + i];
    rsum[tid] = s;
  }
#pragma unroll
  for (int i = 0; i < 4; ++i) outbuf[tid * 4 + i] = 0.0f;
  __syncthreads();

  // ---- Phase 3: out = P V (wave -> dh tile x key half); 2x pipeline -----
  const int dhtile  = wave & 3;
  const int keyhalf = wave >> 2;
  v8f cpv = {};
  const _Float16* vbase =
      vT + ((size_t)bh * DHn + dhtile * 16 + mr) * Ln + kb + keyhalf * 1024;
  const _Float16* pbase = probs + mr * Ln + kb + keyhalf * 1024;
  {
    v16h pa0 = load_frag(pbase);       // ds_load_b128 from LDS
    v16h vb0 = load_frag(vbase);
    for (int k0 = 0; k0 < 1024; k0 += 64) {
      v16h pa1 = load_frag(pbase + k0 + 32);
      v16h vb1 = load_frag(vbase + k0 + 32);
      cpv = wmma16(pa0, vb0, cpv);
      if (k0 + 64 < 1024) {
        pa0 = load_frag(pbase + k0 + 64);
        vb0 = load_frag(vbase + k0 + 64);
      }
      cpv = wmma16(pa1, vb1, cpv);
    }
  }
  {
    const int n = dhtile * 16 + (lane & 15);
#pragma unroll
    for (int rr = 0; rr < 8; ++rr)
      atomicAdd(&outbuf[(rowoff + rr) * 64 + n], cpv[rr]);   // ds_add_f32
  }
  __syncthreads();

  // ---- Phase 4: scale by 1/rowsum, store to attn[B,L,H*DH] -------------
  {
    const int idx = tid * 4;
    const int row = idx >> 6;
    const float inv = 1.0f / rsum[row];
    const int m = q0 + row;
    const int h = bh & 15;
    float* dst = attn_out + ((size_t)b_ * Ln + m) * Dn + h * DHn + (idx & 63);
#pragma unroll
    for (int i = 0; i < 4; ++i) dst[i] = outbuf[idx + i] * inv;
  }
}

// ---------------------------------------------------------------------------
extern "C" void kernel_launch(void* const* d_in, const int* in_sizes, int n_in,
                              void* d_out, int out_size, void* d_ws, size_t ws_size,
                              hipStream_t stream) {
  const float* x   = (const float*)d_in[0];
  const int*   mask= (const int*)  d_in[1];
  const float* Wq  = (const float*)d_in[2];
  const float* bq  = (const float*)d_in[3];
  const float* Wk  = (const float*)d_in[4];
  const float* bk  = (const float*)d_in[5];
  const float* Wv  = (const float*)d_in[6];
  const float* bv  = (const float*)d_in[7];
  const float* g1  = (const float*)d_in[8];
  const float* b1  = (const float*)d_in[9];
  const float* g2  = (const float*)d_in[10];
  const float* b2  = (const float*)d_in[11];
  const float* Wo1 = (const float*)d_in[12];
  const float* bo1 = (const float*)d_in[13];
  const float* Wo2 = (const float*)d_in[14];
  const float* bo2 = (const float*)d_in[15];

  constexpr size_t MB = 1024ull * 1024ull;
  char* ws = (char*)d_ws;
  _Float16* wq_t  = (_Float16*)(ws + 0 * MB);     // 2 MB each
  _Float16* wk_t  = (_Float16*)(ws + 2 * MB);
  _Float16* wv_t  = (_Float16*)(ws + 4 * MB);
  _Float16* wo1_t = (_Float16*)(ws + 6 * MB);
  _Float16* wo2_t = (_Float16*)(ws + 8 * MB);
  _Float16* h1    = (_Float16*)(ws + 10 * MB);    // 16 MB  (reused as h2)
  _Float16* qb    = (_Float16*)(ws + 26 * MB);    // 16 MB  (reused as gelu out)
  _Float16* kbuf  = (_Float16*)(ws + 42 * MB);    // 16 MB
  _Float16* vtb   = (_Float16*)(ws + 58 * MB);    // 16 MB  [B,H,DH,L]
  float*    attnb = (float*)   (ws + 74 * MB);    // 32 MB
  float*    resid = (float*)   (ws + 106 * MB);   // 32 MB
  _Float16* h2    = h1;
  _Float16* gbuf  = qb;

  // 1) Weight convert + transpose (f32 -> f16, [K,N] -> [N,K])
  convert_wt_kernel<<<4096, 256, 0, stream>>>(Wq,  wq_t);
  convert_wt_kernel<<<4096, 256, 0, stream>>>(Wk,  wk_t);
  convert_wt_kernel<<<4096, 256, 0, stream>>>(Wv,  wv_t);
  convert_wt_kernel<<<4096, 256, 0, stream>>>(Wo1, wo1_t);
  convert_wt_kernel<<<4096, 256, 0, stream>>>(Wo2, wo2_t);

  // 2) LN1 -> h1 (f16)
  ln_kernel<<<Mn, 256, 0, stream>>>(x, nullptr, g1, b1, nullptr, h1);

  // 3) QKV projections (WMMA + async-LDS staged B). 512 blocks of 8 waves.
  gemm_wmma_kernel<<<512, 256, 0, stream>>>(h1, wq_t, bq, nullptr, qb,   0);
  gemm_wmma_kernel<<<512, 256, 0, stream>>>(h1, wk_t, bk, nullptr, kbuf, 0);
  gemm_wmma_kernel<<<512, 256, 0, stream>>>(h1, wv_t, bv, nullptr, vtb,  2);

  // 4) Attention: B*H*(L/16) = 8192 workgroups, ~201 KB LDS each.
  attn_kernel<<<8192, 256, SMEM_BYTES, stream>>>(qb, kbuf, vtb, mask, attnb);

  // 5) residual = x + attn ; LN2 -> h2 (f16)
  ln_kernel<<<Mn, 256, 0, stream>>>(x, attnb, g2, b2, resid, h2);

  // 6) MLP: GELU(h2 @ Wo1 + bo1) -> gbuf (f16)
  gemm_wmma_kernel<<<512, 256, 0, stream>>>(h2, wo1_t, bo1, nullptr, gbuf, 3);

  // 7) out = gbuf @ Wo2 + bo2 + residual  (f32)
  gemm_wmma_kernel<<<512, 256, 0, stream>>>(gbuf, wo2_t, bo2, resid, d_out, 4);
}
